// SelfAttentionV2_36129264893953
// MI455X (gfx1250) — compile-verified, matrix operand order checked by reading, not run on hardware
//
#include <hip/hip_runtime.h>
#include <math.h>

// SelfAttention: S=8192, D=1024, fp32 in/out, bf16 WMMA compute (f32 accum).
#define S_SEQ 8192
#define D_DIM 1024

// LDS row paddings (keep 16B alignment, break 64-bank periodicity)
#define QS_LD 1032   // 1024 + 8  -> row stride 2064B = 516 dwords (516%64=4)
#define SS_LD 132    // 128 + 4   -> row stride 528B  = 132 dwords (132%64=4)
#define PS_LD 136    // 128 + 8   -> row stride 272B  = 68 dwords  (68%64=4)

typedef __attribute__((ext_vector_type(8)))  __bf16 v8bf;
typedef __attribute__((ext_vector_type(16))) __bf16 v16bf;
typedef __attribute__((ext_vector_type(8)))  float  v8f;

// ---------------------------------------------------------------------------
// Fragment loader for 16-bit A/B operands of V_WMMA_F32_16X16X32_BF16.
// A (16x32, MxK): lane l (l<16) holds row M=l, K={k0..k0+7, k0+16..k0+23};
//                 lane l (l>=16) holds row M=l-16, K={k0+8..k0+15, k0+24..k0+31}.
// B (32x16, KxN) mirrors A with N in place of M: column n of B is a K-vector,
// so "row n of a row-major [N][K] matrix" is exactly the B fragment source.
// Two contiguous 16-byte loads per lane.
// ---------------------------------------------------------------------------
__device__ __forceinline__ v16bf load_frag_rows(const __bf16* base, int ld,
                                                int k0, int lane) {
  const int row = lane & 15;
  const int kk  = k0 + ((lane >> 4) << 3);      // +8 for upper half-wave
  const __bf16* p = base + (size_t)row * ld + kk;
  v8bf lo = *(const v8bf*)(p);
  v8bf hi = *(const v8bf*)(p + 16);
  return __builtin_shufflevector(lo, hi, 0, 1, 2, 3, 4, 5, 6, 7,
                                         8, 9, 10, 11, 12, 13, 14, 15);
}

__device__ __forceinline__ v8f wmma_bf16(v16bf a, v16bf b, v8f c) {
  // (neg_a, A, neg_b, B, c_mod, C, reuse_a, reuse_b)
  return __builtin_amdgcn_wmma_f32_16x16x32_bf16(false, a, false, b,
                                                 (short)0, c, false, false);
}

// ---------------------------------------------------------------------------
// Kernel 1: fp32 -> bf16 conversion (grid-stride).
// ---------------------------------------------------------------------------
__global__ __launch_bounds__(256) void f32_to_bf16_kernel(
    const float* __restrict__ in, __bf16* __restrict__ out, int n) {
  int i = blockIdx.x * blockDim.x + threadIdx.x;
  int stride = gridDim.x * blockDim.x;
  for (; i < n; i += stride) out[i] = (__bf16)in[i];
}

// ---------------------------------------------------------------------------
// Kernel 2: Y[M,N] = X[M,K] @ W[N,K]^T  (all bf16, f32 accum, bf16 out).
// Block = 8 waves covering 128(M) x 64(N); each wave a 32x32 tile (2x2 WMMA).
// ---------------------------------------------------------------------------
__global__ __launch_bounds__(256) void gemm_bf16_xwT(
    const __bf16* __restrict__ X, const __bf16* __restrict__ W,
    __bf16* __restrict__ Y, int M, int N, int K) {
  const int lane = threadIdx.x & 31;
  const int w    = threadIdx.x >> 5;
  const int m0   = blockIdx.y * 128 + (w & 3) * 32;
  const int n0   = blockIdx.x * 64 + (w >> 2) * 32;

  v8f acc[2][2] = {};
  const __bf16* xa0 = X + (size_t)m0 * K;
  const __bf16* xa1 = X + (size_t)(m0 + 16) * K;
  const __bf16* wb0 = W + (size_t)n0 * K;
  const __bf16* wb1 = W + (size_t)(n0 + 16) * K;

  for (int k0 = 0; k0 < K; k0 += 32) {
    v16bf a0 = load_frag_rows(xa0, K, k0, lane);
    v16bf a1 = load_frag_rows(xa1, K, k0, lane);
    v16bf b0 = load_frag_rows(wb0, K, k0, lane);
    v16bf b1 = load_frag_rows(wb1, K, k0, lane);
    acc[0][0] = wmma_bf16(a0, b0, acc[0][0]);
    acc[0][1] = wmma_bf16(a0, b1, acc[0][1]);
    acc[1][0] = wmma_bf16(a1, b0, acc[1][0]);
    acc[1][1] = wmma_bf16(a1, b1, acc[1][1]);
  }

#pragma unroll
  for (int mt = 0; mt < 2; mt++) {
#pragma unroll
    for (int nt = 0; nt < 2; nt++) {
#pragma unroll
      for (int r = 0; r < 8; r++) {
        int row = m0 + mt * 16 + r + ((lane >> 4) << 3);
        int col = n0 + nt * 16 + (lane & 15);
        Y[(size_t)row * N + col] = (__bf16)acc[mt][nt][r];
      }
    }
  }
}

// ---------------------------------------------------------------------------
// Kernel 3: bf16 transpose  in[R][C] -> out[C][R]  (LDS-tiled, coalesced).
// ---------------------------------------------------------------------------
__global__ __launch_bounds__(256) void transpose_bf16_kernel(
    const __bf16* __restrict__ in, __bf16* __restrict__ out, int R, int C) {
  __shared__ __bf16 tile[32][33];
  const int c0 = blockIdx.x * 32;
  const int r0 = blockIdx.y * 32;
  for (int y = threadIdx.y; y < 32; y += 8)
    tile[y][threadIdx.x] = in[(size_t)(r0 + y) * C + c0 + threadIdx.x];
  __syncthreads();
  for (int y = threadIdx.y; y < 32; y += 8)
    out[(size_t)(c0 + y) * R + r0 + threadIdx.x] = tile[threadIdx.x][y];
}

// ---------------------------------------------------------------------------
// Kernel 4: flash attention.
//   Block = 256 threads (8 waves), 32 query rows, key blocks of BN=128.
//   Q tile (32x1024 bf16, padded rows) staged once in LDS, reused for all 64
//   key blocks (320 KB LDS/WGP makes this cheap).
//   S stage : wave w computes the 32x16 score strip for keys w*16..w*16+15
//             (each B fragment feeds two WMMAs; K rows disjoint across waves).
//   Softmax : all 256 threads; thread = (row, 16-col segment) with LDS
//             two-level max/sum reduction.
//   PV stage: wave w owns output dims [w*128, w*128+128) -> 16 f32
//             accumulator tiles (128 VGPRs) held across all key blocks.
// ---------------------------------------------------------------------------
__global__ __launch_bounds__(256) void flash_attn_kernel(
    const __bf16* __restrict__ Q, const __bf16* __restrict__ Kmat,
    const __bf16* __restrict__ Vt, float* __restrict__ Out) {
  __shared__ __bf16 Qs[32][QS_LD];       // ~64.5 KB
  __shared__ float  Ss[32][SS_LD];       // ~16.5 KB raw scores
  __shared__ __bf16 Ps[32][PS_LD];       // ~8.5 KB  exp(scores) bf16
  __shared__ float  red[32][8];          // partial max/sum
  __shared__ float  mrow[32], lrow[32], frow[32];

  const int   tid  = threadIdx.x;
  const int   lane = tid & 31;
  const int   w    = tid >> 5;
  const int   qb   = blockIdx.x * 32;
  const int   d0   = w * 128;            // PV output-dim slice
  const int   r    = tid & 31;           // softmax row
  const int   seg  = tid >> 5;           // softmax col segment (16 cols)
  const float SCALE = 0.03125f;          // 1/sqrt(1024)
  const float LOG2E = 1.4426950408889634f;

  {  // stage Q tile into LDS (16B copies, row-padded destination)
    const v8bf* src = (const v8bf*)(Q + (size_t)qb * D_DIM);
    for (int i = tid; i < 32 * (D_DIM / 8); i += 256) {
      int row = i / (D_DIM / 8);
      int cs  = i % (D_DIM / 8);
      *(v8bf*)(&Qs[row][cs * 8]) = src[i];
    }
  }
  if (tid < 32) { mrow[tid] = -INFINITY; lrow[tid] = 0.f; }
  __syncthreads();

  v8f oacc[2][8] = {};

  for (int kb = 0; kb < S_SEQ; kb += 128) {
    // ---- prefetch next key block's K rows into cache (global_prefetch_b8)
    if (kb + 128 < S_SEQ) {
      const __bf16* pf =
          Kmat + (size_t)(kb + 128 + w * 16 + (lane & 15)) * D_DIM;
      __builtin_prefetch(pf, 0, 1);
    }

    // ---- S strip: Q(32x1024) x K^T(1024x16) -> 64 WMMAs per wave ----
    v8f sacc0 = {}, sacc1 = {};
    const __bf16* kbase = Kmat + (size_t)(kb + w * 16) * D_DIM;
    for (int k0 = 0; k0 < D_DIM; k0 += 32) {
      v16bf a0 = load_frag_rows(&Qs[0][0],  QS_LD, k0, lane);
      v16bf a1 = load_frag_rows(&Qs[16][0], QS_LD, k0, lane);
      v16bf b  = load_frag_rows(kbase, D_DIM, k0, lane);
      sacc0 = wmma_bf16(a0, b, sacc0);
      sacc1 = wmma_bf16(a1, b, sacc1);
    }
#pragma unroll
    for (int rr = 0; rr < 8; rr++) {
      int mr = rr + ((lane >> 4) << 3);
      int cc = w * 16 + (lane & 15);
      Ss[mr][cc]      = sacc0[rr] * SCALE;
      Ss[mr + 16][cc] = sacc1[rr] * SCALE;
    }
    __syncthreads();

    // ---- softmax, all 256 threads: partial row max ----
    {
      float mx = -INFINITY;
#pragma unroll
      for (int j = 0; j < 16; j++) mx = fmaxf(mx, Ss[r][seg * 16 + j]);
      red[r][seg] = mx;
    }
    __syncthreads();
    if (tid < 32) {  // combine: new running max + rescale factor
      float mold = mrow[r];
      float mx = mold;
#pragma unroll
      for (int s = 0; s < 8; s++) mx = fmaxf(mx, red[r][s]);
      frow[r] = exp2f((mold - mx) * LOG2E);  // 0 on first block
      mrow[r] = mx;
    }
    __syncthreads();
    {  // exp + partial row sum
      float mx = mrow[r];
      float sum = 0.f;
#pragma unroll
      for (int j = 0; j < 16; j++) {
        float p = exp2f((Ss[r][seg * 16 + j] - mx) * LOG2E);
        Ps[r][seg * 16 + j] = (__bf16)p;
        sum += p;
      }
      red[r][seg] = sum;
    }
    __syncthreads();
    if (tid < 32) {  // combine row sums into running denominator
      float s = 0.f;
#pragma unroll
      for (int sg = 0; sg < 8; sg++) s += red[r][sg];
      lrow[r] = lrow[r] * frow[r] + s;
    }

    // ---- rescale running O by exp(m_old - m_new) ----
#pragma unroll
    for (int rt = 0; rt < 2; rt++) {
#pragma unroll
      for (int rr = 0; rr < 8; rr++) {
        float f = frow[rt * 16 + rr + ((lane >> 4) << 3)];
#pragma unroll
        for (int ct = 0; ct < 8; ct++) oacc[rt][ct][rr] *= f;
      }
    }

    // ---- O += P(32x128) x V(128x128-slice): 64 WMMAs per wave ----
#pragma unroll
    for (int kt = 0; kt < 4; kt++) {
      v16bf a0 = load_frag_rows(&Ps[0][0],  PS_LD, kt * 32, lane);
      v16bf a1 = load_frag_rows(&Ps[16][0], PS_LD, kt * 32, lane);
#pragma unroll
      for (int ct = 0; ct < 8; ct++) {
        // Vt is [D][S]: column n of B (= fixed dim, over keys) is contiguous.
        const __bf16* vbase = Vt + (size_t)(d0 + ct * 16) * S_SEQ + kb;
        v16bf b = load_frag_rows(vbase, S_SEQ, kt * 32, lane);
        oacc[0][ct] = wmma_bf16(a0, b, oacc[0][ct]);
        oacc[1][ct] = wmma_bf16(a1, b, oacc[1][ct]);
      }
    }
    __syncthreads();   // protect Ss/Ps/frow before next block overwrites
  }

  // ---- normalize by row sums and store fp32 ----
#pragma unroll
  for (int rt = 0; rt < 2; rt++) {
#pragma unroll
    for (int rr = 0; rr < 8; rr++) {
      int row = rt * 16 + rr + ((lane >> 4) << 3);
      float inv = 1.f / lrow[row];
#pragma unroll
      for (int ct = 0; ct < 8; ct++) {
        int col = d0 + ct * 16 + (lane & 15);
        Out[(size_t)(qb + row) * D_DIM + col] = oacc[rt][ct][rr] * inv;
      }
    }
  }
}

// ---------------------------------------------------------------------------
// Host-side launcher.
// ---------------------------------------------------------------------------
extern "C" void kernel_launch(void* const* d_in, const int* in_sizes, int n_in,
                              void* d_out, int out_size, void* d_ws, size_t ws_size,
                              hipStream_t stream) {
  (void)in_sizes; (void)n_in; (void)out_size; (void)ws_size;
  const float* x  = (const float*)d_in[0];
  const float* Wq = (const float*)d_in[1];
  const float* Wk = (const float*)d_in[2];
  const float* Wv = (const float*)d_in[3];
  float* out = (float*)d_out;

  // Workspace layout (bf16), ~86 MB total.
  __bf16* Xb  = (__bf16*)d_ws;                       // [S][D]
  __bf16* Wqb = Xb  + (size_t)S_SEQ * D_DIM;         // [D][D]
  __bf16* Wkb = Wqb + (size_t)D_DIM * D_DIM;
  __bf16* Wvb = Wkb + (size_t)D_DIM * D_DIM;
  __bf16* Qm  = Wvb + (size_t)D_DIM * D_DIM;         // [S][D]
  __bf16* Km  = Qm  + (size_t)S_SEQ * D_DIM;         // [S][D]
  __bf16* Vr  = Km  + (size_t)S_SEQ * D_DIM;         // [S][D]
  __bf16* Vtm = Vr  + (size_t)S_SEQ * D_DIM;         // [D][S]

  f32_to_bf16_kernel<<<2048, 256, 0, stream>>>(x,  Xb,  S_SEQ * D_DIM);
  f32_to_bf16_kernel<<<512,  256, 0, stream>>>(Wq, Wqb, D_DIM * D_DIM);
  f32_to_bf16_kernel<<<512,  256, 0, stream>>>(Wk, Wkb, D_DIM * D_DIM);
  f32_to_bf16_kernel<<<512,  256, 0, stream>>>(Wv, Wvb, D_DIM * D_DIM);

  dim3 ggrid(D_DIM / 64, S_SEQ / 128);
  gemm_bf16_xwT<<<ggrid, 256, 0, stream>>>(Xb, Wqb, Qm, S_SEQ, D_DIM, D_DIM);
  gemm_bf16_xwT<<<ggrid, 256, 0, stream>>>(Xb, Wkb, Km, S_SEQ, D_DIM, D_DIM);
  gemm_bf16_xwT<<<ggrid, 256, 0, stream>>>(Xb, Wvb, Vr, S_SEQ, D_DIM, D_DIM);

  dim3 tgrid(D_DIM / 32, S_SEQ / 32), tblock(32, 8);
  transpose_bf16_kernel<<<tgrid, tblock, 0, stream>>>(Vr, Vtm, S_SEQ, D_DIM);

  flash_attn_kernel<<<S_SEQ / 32, 256, 0, stream>>>(Qm, Km, Vtm, out);
}
